// HybridRNABindingSiteModel_11519102288369
// MI455X (gfx1250) — matrix operand order repaired and checked by reading, not compiled, not run on hardware
//
#include <hip/hip_runtime.h>
#include <math.h>

// ---------------- problem constants ----------------
#define NNODES 50000
#define NEDGES 800000
#define NTOT   (NEDGES + NNODES)   // edges + self loops
#define NGRP   200
#define HD     128
#define EPS_F  1e-5f

typedef __bf16 v16bf __attribute__((ext_vector_type(16)));
typedef __bf16 v8bf  __attribute__((ext_vector_type(8)));
typedef __bf16 v4bf  __attribute__((ext_vector_type(4)));
typedef float  v8f   __attribute__((ext_vector_type(8)));

__device__ __forceinline__ __bf16 f2bf(float f) {
  unsigned u = __float_as_uint(f);
  unsigned r = (u + 0x7FFFu + ((u >> 16) & 1u)) >> 16;   // round-to-nearest-even
  unsigned short h = (unsigned short)r;
  return __builtin_bit_cast(__bf16, h);
}

// ---------------------------------------------------------------------------
// Generic bf16 WMMA GEMM: C[M,Nc] = concat(A0,A1)[M,K] * W[K,Nc] + bias
// A element (m,k) = (k < K1) ? A0[m*K1+k] : A1[m*(K-K1)+(k-K1)]
// Block: 256 threads = 8 waves; block computes 64 rows x 128 cols.
// Wave w owns col tile w (16 cols) and 4 row tiles -> 4x v_wmma_f32_16x16x32_bf16
// per K-step of 32. LDS tiles are laid out so both A and B fragments are
// contiguous 16B-aligned runs per lane (ds_load_b128 paths). Interior tiles
// take a uniform, unguarded staging fast path (straight global_load_b128 /
// ds_store_b64 code); only boundary tiles pay for per-element guards.
// ---------------------------------------------------------------------------
__global__ __launch_bounds__(256)
void gemm_wmma_bf16(const float* __restrict__ A0, const float* __restrict__ A1,
                    int K1, int K, const float* __restrict__ W,
                    const float* __restrict__ bias, float* __restrict__ C,
                    int M, int Nc)
{
  // column-major W tile: sW2[col][k], row stride 64B -> lane's 16 B-values are
  // 32 contiguous bytes at a 32B-aligned address.
  __shared__ __align__(32) __bf16 sW2[128][32];
  // row-major A tile: sA[row][k], row stride 64B.
  __shared__ __align__(32) __bf16 sA[64][32];

  const int tid   = threadIdx.x;
  const int wave  = tid >> 5;
  const int lane  = tid & 31;
  const int lhalf = lane >> 4;     // 0: lanes 0-15, 1: lanes 16-31
  const int lmod  = lane & 15;
  const int r0    = blockIdx.x * 64;
  const int KA1   = K - K1;
  const bool rows_full = (r0 + 64 <= M);

  v8f acc[4];
  const v8f vzero = {0.f,0.f,0.f,0.f,0.f,0.f,0.f,0.f};
#pragma unroll
  for (int rt = 0; rt < 4; ++rt) acc[rt] = vzero;

  for (int k0 = 0; k0 < K; k0 += 32) {
    // ================= stage W tile =================
    if (k0 + 32 <= K && Nc == 128) {
      // uniform fast path: unguarded coalesced float4 loads
#pragma unroll
      for (int it = 0; it < 4; ++it) {
        int i  = tid + it * 256;       // 0..1023
        int kk = i >> 5;               // 0..31
        int c4 = (i & 31) << 2;        // 0..124
        float4 v = *(const float4*)(W + (size_t)(k0 + kk) * 128 + c4);
        sW2[c4 + 0][kk] = f2bf(v.x);
        sW2[c4 + 1][kk] = f2bf(v.y);
        sW2[c4 + 2][kk] = f2bf(v.z);
        sW2[c4 + 3][kk] = f2bf(v.w);
      }
    } else {
      // boundary path: clamped addresses + zero-select
#pragma unroll
      for (int it = 0; it < 4; ++it) {
        int i   = tid + it * 256;
        int kk  = i >> 5;
        int c4  = (i & 31) << 2;
        int k   = k0 + kk;
        int kc  = k < K ? k : K - 1;
        int c4c = c4 < Nc ? c4 : 0;
        float4 v = *(const float4*)(W + (size_t)kc * Nc + c4c);
        if (!((k < K) && (c4 < Nc))) { v.x = 0.f; v.y = 0.f; v.z = 0.f; v.w = 0.f; }
        sW2[c4 + 0][kk] = f2bf(v.x);
        sW2[c4 + 1][kk] = f2bf(v.y);
        sW2[c4 + 2][kk] = f2bf(v.z);
        sW2[c4 + 3][kk] = f2bf(v.w);
      }
    }

    // ================= stage A tile =================
    const bool inA0 = (k0 + 32 <= K1);
    const bool inA1 = (k0 >= K1);
    if (rows_full && (k0 + 32 <= K) && (inA0 || inA1)) {
      // uniform fast path: tile fully inside one region -> float4 loads,
      // packed 8B LDS stores, plus prefetch of the next K-tile.
      const float* base  = inA0 ? (A0 + k0) : (A1 + (k0 - K1));
      const int    strid = inA0 ? K1 : KA1;
      const int    regend = inA0 ? K1 : K;
      const bool   pf = (k0 + 64 <= regend);
#pragma unroll
      for (int it = 0; it < 2; ++it) {
        int i  = tid + it * 256;       // 0..511
        int rr = i >> 3;               // 0..63
        int k4 = (i & 7) << 2;         // 0..28
        const float* p = base + (size_t)(r0 + rr) * strid + k4;
        float4 v = *(const float4*)p;
        if (pf) __builtin_prefetch(p + 32, 0, 3);   // global_prefetch_b8
        v4bf pk = { f2bf(v.x), f2bf(v.y), f2bf(v.z), f2bf(v.w) };
        *(v4bf*)&sA[rr][k4] = pk;
      }
    } else {
      // boundary path (last row block / K tail / concat straddle)
#pragma unroll
      for (int it = 0; it < 8; ++it) {
        int i  = tid + it * 256;       // 0..2047
        int rr = i >> 5, kk = i & 31;
        int r = r0 + rr, k = k0 + kk;
        int rc = r < M ? r : M - 1;
        int kc = k < K ? k : K - 1;
        const float* p = (kc < K1) ? (A0 + (size_t)rc * K1 + kc)
                                   : (A1 + (size_t)rc * KA1 + (kc - K1));
        float v = *p;
        if (!((r < M) && (k < K))) v = 0.f;
        sA[rr][kk] = f2bf(v);
      }
    }
    __syncthreads();

    // ---- B fragment: 32x16 bf16. lanes 0-15 hold K=0..15, lanes 16-31 K=16..31
    const int col = (wave << 4) + lmod;
    v8bf b0 = *(const v8bf*)&sW2[col][(lhalf << 4)];
    v8bf b1 = *(const v8bf*)&sW2[col][(lhalf << 4) + 8];
    v16bf bfrag;
#pragma unroll
    for (int j = 0; j < 8; ++j) { bfrag[j] = b0[j]; bfrag[8 + j] = b1[j]; }

#pragma unroll
    for (int rt = 0; rt < 4; ++rt) {
      // ---- A fragment: 16x32 bf16. lanes 0-15 hold K=0..15, lanes 16-31 K=16..31
      const int ar = (rt << 4) + lmod;
      v8bf a0 = *(const v8bf*)&sA[ar][(lhalf << 4)];
      v8bf a1 = *(const v8bf*)&sA[ar][(lhalf << 4) + 8];
      v16bf afrag;
#pragma unroll
      for (int j = 0; j < 8; ++j) { afrag[j] = a0[j]; afrag[8 + j] = a1[j]; }

      acc[rt] = __builtin_amdgcn_wmma_f32_16x16x32_bf16(
          /*neg_a=*/false, afrag, /*neg_b=*/false, bfrag,
          /*c_mod=*/(short)0, acc[rt], /*reuse_a=*/false, /*reuse_b=*/false);
    }
    __syncthreads();
  }

  // ---- epilogue: D layout VGPR i -> (M = i + 8*lhalf, N = lmod)
  const int col = (wave << 4) + lmod;
  if (col < Nc) {
    const float bv = bias ? bias[col] : 0.f;
    if (rows_full) {
#pragma unroll
      for (int rt = 0; rt < 4; ++rt) {
#pragma unroll
        for (int i = 0; i < 8; ++i) {
          int r = r0 + (rt << 4) + (lhalf << 3) + i;
          C[(size_t)r * Nc + col] = acc[rt][i] + bv;
        }
      }
    } else {
#pragma unroll
      for (int rt = 0; rt < 4; ++rt) {
#pragma unroll
        for (int i = 0; i < 8; ++i) {
          int r = r0 + (rt << 4) + (lhalf << 3) + i;
          if (r < M) C[(size_t)r * Nc + col] = acc[rt][i] + bv;
        }
      }
    }
  }
}

// ---------------------------------------------------------------------------
// LayerNorm (+ReLU) over rows of width w
// ---------------------------------------------------------------------------
__global__ void ln_relu_kernel(const float* __restrict__ in, float* __restrict__ outp,
                               const float* __restrict__ g, const float* __restrict__ b,
                               int n, int w)
{
  int i = blockIdx.x * blockDim.x + threadIdx.x;
  if (i >= n) return;
  const float* x = in + (size_t)i * w;
  float m = 0.f;
  for (int d = 0; d < w; ++d) m += x[d];
  m /= (float)w;
  float v = 0.f;
  for (int d = 0; d < w; ++d) { float t = x[d] - m; v += t * t; }
  v /= (float)w;
  float rs = rsqrtf(v + EPS_F);
  float* o = outp + (size_t)i * w;
  for (int d = 0; d < w; ++d) {
    float y = (x[d] - m) * rs * g[d] + b[d];
    o[d] = y > 0.f ? y : 0.f;
  }
}

// per-node attention score dots: s_src = xw . att_src ; s_dst = xw . att_dst
__global__ void scores_kernel(const float* __restrict__ xw,
                              const float* __restrict__ as, const float* __restrict__ ad,
                              float* __restrict__ ssrc, float* __restrict__ sdst)
{
  int i = blockIdx.x * blockDim.x + threadIdx.x;
  if (i >= NNODES) return;
  const float* x = xw + (size_t)i * HD;
  float a = 0.f, b = 0.f;
  for (int d = 0; d < HD; ++d) { float xv = x[d]; a += xv * as[d]; b += xv * ad[d]; }
  ssrc[i] = a;
  sdst[i] = b;
}

__global__ void fill_kernel(float* __restrict__ p, int n, float v)
{
  int i = blockIdx.x * blockDim.x + threadIdx.x;
  if (i < n) p[i] = v;
}

// leaky-relu edge score + per-dst float atomic max (int/uint ordering trick)
__global__ void edge_max_kernel(const int* __restrict__ ei,
                                const float* __restrict__ ssrc, const float* __restrict__ sdst,
                                float* __restrict__ ebuf, int* __restrict__ mbits)
{
  int i = blockIdx.x * blockDim.x + threadIdx.x;
  if (i >= NTOT) return;
  int s = (i < NEDGES) ? ei[i]          : (i - NEDGES);
  int t = (i < NEDGES) ? ei[NEDGES + i] : (i - NEDGES);
  float sc = ssrc[s] + sdst[t];
  sc = sc > 0.f ? sc : 0.2f * sc;
  ebuf[i] = sc;
  if (sc >= 0.f) atomicMax(mbits + t, __float_as_int(sc));
  else           atomicMin((unsigned int*)(mbits + t), __float_as_uint(sc));
}

__global__ void edge_exp_kernel(const int* __restrict__ ei, float* __restrict__ ebuf,
                                const int* __restrict__ mbits, float* __restrict__ sums)
{
  int i = blockIdx.x * blockDim.x + threadIdx.x;
  if (i >= NTOT) return;
  int t = (i < NEDGES) ? ei[NEDGES + i] : (i - NEDGES);
  float e = __expf(ebuf[i] - __int_as_float(mbits[t]));
  ebuf[i] = e;
  atomicAdd(sums + t, e);
}

__global__ void bias_init_kernel(float* __restrict__ agg, const float* __restrict__ bias)
{
  int i = blockIdx.x * blockDim.x + threadIdx.x;
  if (i < NNODES * HD) agg[i] = bias[i & (HD - 1)];
}

// wave-per-edge scatter: dst += alpha * xw[src]  (each lane: 4 channels)
__global__ void edge_agg_kernel(const int* __restrict__ ei, const float* __restrict__ ebuf,
                                const float* __restrict__ sums, const float* __restrict__ xw,
                                float* __restrict__ agg)
{
  int gid = blockIdx.x * blockDim.x + threadIdx.x;
  int w = gid >> 5;
  if (w >= NTOT) return;
  int lane = gid & 31;
  int s = (w < NEDGES) ? ei[w]          : (w - NEDGES);
  int t = (w < NEDGES) ? ei[NEDGES + w] : (w - NEDGES);
  float alpha = ebuf[w] / (sums[t] + 1e-16f);
  const float4 x = *(const float4*)(xw + (size_t)s * HD + lane * 4);
  float* o = agg + (size_t)t * HD + lane * 4;
  atomicAdd(o + 0, alpha * x.x);
  atomicAdd(o + 1, alpha * x.y);
  atomicAdd(o + 2, alpha * x.z);
  atomicAdd(o + 3, alpha * x.w);
}

// ----- GraphNorm (3 passes; sums via atomics into [G,H]) -----
__global__ void gn_pass1(const float* __restrict__ x, const int* __restrict__ batch,
                         float* __restrict__ gsum, float* __restrict__ gcnt)
{
  int gid = blockIdx.x * blockDim.x + threadIdx.x;
  int n = gid >> 5;
  if (n >= NNODES) return;
  int lane = gid & 31;
  int g = batch[n];
  const float* xr = x + (size_t)n * HD + lane * 4;
  float* sp = gsum + (size_t)g * HD + lane * 4;
  atomicAdd(sp + 0, xr[0]);
  atomicAdd(sp + 1, xr[1]);
  atomicAdd(sp + 2, xr[2]);
  atomicAdd(sp + 3, xr[3]);
  if (lane == 0) atomicAdd(gcnt + g, 1.f);
}

__global__ void gn_pass2(const float* __restrict__ x, const int* __restrict__ batch,
                         const float* __restrict__ gsum, const float* __restrict__ gcnt,
                         const float* __restrict__ ms, float* __restrict__ tmp,
                         float* __restrict__ gsq)
{
  int gid = blockIdx.x * blockDim.x + threadIdx.x;
  int n = gid >> 5;
  if (n >= NNODES) return;
  int lane = gid & 31;
  int g = batch[n];
  float c = gcnt[g]; c = c < 1.f ? 1.f : c;
#pragma unroll
  for (int j = 0; j < 4; ++j) {
    int d = lane * 4 + j;
    float mean = gsum[(size_t)g * HD + d] / c;
    float o = x[(size_t)n * HD + d] - ms[d] * mean;
    tmp[(size_t)n * HD + d] = o;
    atomicAdd(gsq + (size_t)g * HD + d, o * o);
  }
}

__global__ void gn_pass3(const float* __restrict__ tmp, const int* __restrict__ batch,
                         const float* __restrict__ gsq, const float* __restrict__ gcnt,
                         const float* __restrict__ wv, const float* __restrict__ bv,
                         float* __restrict__ outp, int relu)
{
  int gid = blockIdx.x * blockDim.x + threadIdx.x;
  int n = gid >> 5;
  if (n >= NNODES) return;
  int lane = gid & 31;
  int g = batch[n];
  float c = gcnt[g]; c = c < 1.f ? 1.f : c;
#pragma unroll
  for (int j = 0; j < 4; ++j) {
    int d = lane * 4 + j;
    float sd = sqrtf(gsq[(size_t)g * HD + d] / c + EPS_F);
    float o = wv[d] * tmp[(size_t)n * HD + d] / sd + bv[d];
    if (relu) o = o > 0.f ? o : 0.f;
    outp[(size_t)n * HD + d] = o;
  }
}

__global__ void add_inplace_kernel(float* __restrict__ a, const float* __restrict__ b, int n)
{
  int i = blockIdx.x * blockDim.x + threadIdx.x;
  if (i < n) a[i] += b[i];
}

__global__ void gate_combine_kernel(const float* __restrict__ g, const float* __restrict__ h1,
                                    const float* __restrict__ h2, float* __restrict__ o, int n)
{
  int i = blockIdx.x * blockDim.x + threadIdx.x;
  if (i < n) {
    float gt = 1.f / (1.f + __expf(-g[i]));
    o[i] = gt * h1[i] + (1.f - gt) * h2[i];
  }
}

__global__ void fc2_kernel(const float* __restrict__ in, const float* __restrict__ w2,
                           const float* __restrict__ b2, float* __restrict__ out)
{
  int i = blockIdx.x * blockDim.x + threadIdx.x;
  if (i >= NNODES) return;
  const float* x = in + (size_t)i * 64;
  float a = b2[0];
  for (int d = 0; d < 64; ++d) a += x[d] * w2[d];
  out[i] = a;
}

// ---------------------------------------------------------------------------
static inline int cdiv_i(int a, int b) { return (a + b - 1) / b; }

static void run_gat(const float* hin, const float* Wg, const float* as, const float* ad,
                    const float* bg, const int* ei, float* xw, float* ssrc, float* sdst,
                    float* mbuf, float* sums, float* ebuf, float* agg, hipStream_t s)
{
  gemm_wmma_bf16<<<cdiv_i(NNODES, 64), 256, 0, s>>>(hin, nullptr, HD, HD, Wg, nullptr, xw, NNODES, HD);
  scores_kernel<<<cdiv_i(NNODES, 256), 256, 0, s>>>(xw, as, ad, ssrc, sdst);
  fill_kernel<<<cdiv_i(NNODES, 256), 256, 0, s>>>(mbuf, NNODES, -INFINITY);
  fill_kernel<<<cdiv_i(NNODES, 256), 256, 0, s>>>(sums, NNODES, 0.f);
  edge_max_kernel<<<cdiv_i(NTOT, 256), 256, 0, s>>>(ei, ssrc, sdst, ebuf, (int*)mbuf);
  edge_exp_kernel<<<cdiv_i(NTOT, 256), 256, 0, s>>>(ei, ebuf, (const int*)mbuf, sums);
  bias_init_kernel<<<cdiv_i(NNODES * HD, 256), 256, 0, s>>>(agg, bg);
  edge_agg_kernel<<<cdiv_i(NTOT * 32, 256), 256, 0, s>>>(ei, ebuf, sums, xw, agg);
}

static void run_gn(const float* x, float* tmp, float* outp, const int* batch,
                   const float* wv, const float* bv, const float* ms,
                   float* gnsum, int relu, hipStream_t s)
{
  // gnsum / gnsq / gncnt are contiguous: zero all at once
  fill_kernel<<<cdiv_i(NGRP * HD * 2 + NGRP, 256), 256, 0, s>>>(gnsum, NGRP * HD * 2 + NGRP, 0.f);
  float* gnsq  = gnsum + NGRP * HD;
  float* gncnt = gnsq + NGRP * HD;
  gn_pass1<<<cdiv_i(NNODES * 32, 256), 256, 0, s>>>(x, batch, gnsum, gncnt);
  gn_pass2<<<cdiv_i(NNODES * 32, 256), 256, 0, s>>>(x, batch, gnsum, gncnt, ms, tmp, gnsq);
  gn_pass3<<<cdiv_i(NNODES * 32, 256), 256, 0, s>>>(tmp, batch, gnsq, gncnt, wv, bv, outp, relu);
}

extern "C" void kernel_launch(void* const* d_in, const int* in_sizes, int n_in,
                              void* d_out, int out_size, void* d_ws, size_t ws_size,
                              hipStream_t stream)
{
  (void)in_sizes; (void)n_in; (void)out_size; (void)ws_size;
  const float* rna    = (const float*)d_in[0];
  const float* ssm    = (const float*)d_in[1];
  const int*   ei     = (const int*)d_in[2];
  const int*   batch  = (const int*)d_in[3];
  const float* W_fuse = (const float*)d_in[4];
  const float* b_fuse = (const float*)d_in[5];
  const float* ln1g   = (const float*)d_in[6];
  const float* ln1b   = (const float*)d_in[7];
  const float* W_gat  = (const float*)d_in[8];
  const float* att_s  = (const float*)d_in[9];
  const float* att_d  = (const float*)d_in[10];
  const float* b_gat  = (const float*)d_in[11];
  const float* gnw    = (const float*)d_in[12];
  const float* gnb    = (const float*)d_in[13];
  const float* gnms   = (const float*)d_in[14];
  const float* W_gate = (const float*)d_in[15];
  const float* b_gate = (const float*)d_in[16];
  const float* W_head = (const float*)d_in[17];
  const float* b_head = (const float*)d_in[18];
  const float* ln2g   = (const float*)d_in[19];
  const float* ln2b   = (const float*)d_in[20];
  const float* W_fc1  = (const float*)d_in[21];
  const float* b_fc1  = (const float*)d_in[22];
  const float* ln3g   = (const float*)d_in[23];
  const float* ln3b   = (const float*)d_in[24];
  const float* W_fc2  = (const float*)d_in[25];
  const float* b_fc2  = (const float*)d_in[26];
  float* out = (float*)d_out;

  float* ws = (float*)d_ws;
  const size_t NH = (size_t)NNODES * HD;
  float* buf_h   = ws;             // h after fuse+LN+ReLU (= h_res)
  float* buf_xw  = ws + NH;        // GAT projection / generic temp
  float* buf_agg = ws + 2 * NH;    // GAT aggregation / generic temp
  float* buf_h1  = ws + 3 * NH;
  float* buf_h2  = ws + 4 * NH;
  float* gnsum   = ws + 5 * NH;                    // [G,H] sum, then sq, then cnt
  float* ssrc    = gnsum + (size_t)NGRP * HD * 2 + NGRP;
  float* sdst    = ssrc + NNODES;
  float* mbuf    = sdst + NNODES;
  float* sums    = mbuf + NNODES;
  float* ebuf    = sums + NNODES;                  // NTOT floats

  const int gemm_grid = cdiv_i(NNODES, 64);

  // 1) fuse: concat(rna, ss) @ W_fuse + b  ->  LN1 -> ReLU -> h (= h_res)
  gemm_wmma_bf16<<<gemm_grid, 256, 0, stream>>>(rna, ssm, 645, 651, W_fuse, b_fuse,
                                                buf_agg, NNODES, HD);
  ln_relu_kernel<<<cdiv_i(NNODES, 256), 256, 0, stream>>>(buf_agg, buf_h, ln1g, ln1b, NNODES, HD);

  // 2) GAT layer 1 + GraphNorm + ReLU -> h1
  run_gat(buf_h, W_gat, att_s, att_d, b_gat, ei, buf_xw, ssrc, sdst, mbuf, sums, ebuf, buf_agg, stream);
  run_gn(buf_agg, buf_xw, buf_h1, batch, gnw, gnb, gnms, gnsum, 1, stream);

  // 3) GAT layer 2 (same weights) + GraphNorm + ReLU, then + h_res -> h2
  run_gat(buf_h1, W_gat, att_s, att_d, b_gat, ei, buf_xw, ssrc, sdst, mbuf, sums, ebuf, buf_agg, stream);
  run_gn(buf_agg, buf_xw, buf_h2, batch, gnw, gnb, gnms, gnsum, 1, stream);
  add_inplace_kernel<<<cdiv_i((int)NH, 256), 256, 0, stream>>>(buf_h2, buf_h, (int)NH);

  // 4) gate = sigmoid(concat(h1,h2) @ W_gate + b); h = gate*h1 + (1-gate)*h2
  gemm_wmma_bf16<<<gemm_grid, 256, 0, stream>>>(buf_h1, buf_h2, HD, 2 * HD, W_gate, b_gate,
                                                buf_xw, NNODES, HD);
  gate_combine_kernel<<<cdiv_i((int)NH, 256), 256, 0, stream>>>(buf_xw, buf_h1, buf_h2, buf_agg, (int)NH);

  // 5) head: h @ W_head + b -> LN2 -> ReLU
  gemm_wmma_bf16<<<gemm_grid, 256, 0, stream>>>(buf_agg, nullptr, HD, HD, W_head, b_head,
                                                buf_xw, NNODES, HD);
  ln_relu_kernel<<<cdiv_i(NNODES, 256), 256, 0, stream>>>(buf_xw, buf_agg, ln2g, ln2b, NNODES, HD);

  // 6) fc1: [N,128] @ [128,64] + b -> LN3 -> ReLU
  gemm_wmma_bf16<<<gemm_grid, 256, 0, stream>>>(buf_agg, nullptr, HD, HD, W_fc1, b_fc1,
                                                buf_xw, NNODES, 64);
  ln_relu_kernel<<<cdiv_i(NNODES, 256), 256, 0, stream>>>(buf_xw, buf_agg, ln3g, ln3b, NNODES, 64);

  // 7) fc2: [N,64] @ [64,1] + b -> out[N]
  fc2_kernel<<<cdiv_i(NNODES, 256), 256, 0, stream>>>(buf_agg, W_fc2, b_fc2, out);
}
